// STAD_Encoder_61186104099243
// MI455X (gfx1250) — compile-verified, matrix operand order checked by reading, not run on hardware
//
#include <hip/hip_runtime.h>
#include <math.h>

#define BB 8
#define TT 24
#define NN 1024
#define FE 8
#define HH 64
#define KS 5
#define THd (TT*HH)            // 1536
#define ROWS (BB*TT*NN)        // 196608
#define SZ ((long)ROWS*HH)     // 12582912
#define KC 128                 // K-chunk staged in LDS by the TDM
#define LDP 136                // padded LDS row stride in bf16 elems (272B = 256B + 16B pad)

typedef __attribute__((ext_vector_type(16))) __bf16 v16bf;
typedef __attribute__((ext_vector_type(8)))  float  v8f;
typedef unsigned int u32x4 __attribute__((ext_vector_type(4)));
typedef int          i32x8 __attribute__((ext_vector_type(8)));
typedef int          i32x4 __attribute__((ext_vector_type(4)));

union FragAB { v16bf v; unsigned int d[8]; unsigned short u[16]; };
union FragC  { v8f   v; float f[8]; };

__device__ __forceinline__ unsigned short f2bf(float x){
  unsigned int u = __float_as_uint(x);
  return (unsigned short)((u + 0x7FFFu + ((u>>16)&1u)) >> 16);
}
__device__ __forceinline__ float bf2f(unsigned short s){
  return __uint_as_float(((unsigned int)s)<<16);
}

// Row-major fragment load: element j of the v16bf maps to K = kBase + 8*g + (j&7) + 16*(j>>3)
__device__ __forceinline__ void load_frag(FragAB& f, const unsigned short* __restrict__ base,
                                          int row, int ldk, int kBase, int g){
  const unsigned short* q = base + (long)row*ldk + kBase + 8*g;
#pragma unroll
  for (int p=0;p<4;++p) f.d[p]   = *(const unsigned int*)(q + 2*p);
#pragma unroll
  for (int p=0;p<4;++p) f.d[4+p] = *(const unsigned int*)(q + 16 + 2*p);
}

__device__ __forceinline__ v8f wmma_bf16(const FragAB& a, const FragAB& b, v8f c){
  return __builtin_amdgcn_wmma_f32_16x16x32_bf16(false, a.v, false, b.v, (short)0, c, false, false);
}

// --- Tensor Data Mover: 2D tile [tile_dim1=64 rows][tile_dim0=128 elems] of bf16,
//     row stride in memory = 1024 elems; TDM inserts 4 DWORDs of LDS padding every
//     64 DWORDs (per 256B row) so LDS rows stride 272B (bank-conflict-free frags).
//     6-arg builtin form (clang-23 / therock-10.0 lane).
__device__ __forceinline__ void tdm_load_tile(unsigned lds_off, const void* gaddr){
  unsigned long long ga = (unsigned long long)gaddr;
  u32x4 g0;
  g0[0] = 1u;                                              // count=1 (valid user D#)
  g0[1] = lds_off;                                         // lds_addr (bytes)
  g0[2] = (unsigned)(ga & 0xFFFFFFFFu);                    // global_addr[31:0]
  g0[3] = (unsigned)((ga >> 32) & 0x01FFFFFFu) | (2u<<30); // global_addr[56:32], type=2
  i32x8 g1;
  g1[0] = (1<<16) | (1<<20) | (5<<22) | (3<<25);  // data_size=2B, pad_en, intrvl=64DW, amt=4DW
  g1[1] = (int)((NN & 0xFFFF) << 16);             // tensor_dim0[15:0] (bits 63:48)
  g1[2] = (int)((NN >> 16) & 0xFFFF)              // tensor_dim0[31:16]
        | ((HH & 0xFFFF) << 16);                  // tensor_dim1[15:0]
  g1[3] = 0 | ((KC & 0xFFFF) << 16);              // tensor_dim1[31:16]=0, tile_dim0=KC
  g1[4] = HH;                                     // tile_dim1=64, tile_dim2=0
  g1[5] = NN;                                     // tensor_dim0_stride[31:0]
  g1[6] = 0;                                      // stride hi + dim1_stride lo
  g1[7] = 0;
  i32x4 z4 = {0,0,0,0};
  i32x8 z8 = {0,0,0,0,0,0,0,0};
  __builtin_amdgcn_tensor_load_to_lds(g0, g1, z4, z4, z8, 0);
}

// ---------------- weight convert (f32 [mats][K][64] -> bf16 [mats][64][K]) -------------
__global__ __launch_bounds__(256) void k_convert_tr(const float* __restrict__ src,
    unsigned short* __restrict__ dst, int K, long total){
  long gid = (long)blockIdx.x*256 + threadIdx.x;
  if (gid >= total) return;
  long mk = gid >> 6;
  int  c  = (int)(gid & 63);
  long m  = mk / K;
  int  k  = (int)(mk - m*K);
  dst[(m*HH)*K + (long)c*K + k] = f2bf(src[gid]);
}

// ---------------- plain f32 -> bf16 convert -------------------------------------------
__global__ __launch_bounds__(256) void k_convert(const float* __restrict__ src,
    unsigned short* __restrict__ dst, long total){
  long gid = (long)blockIdx.x*256 + threadIdx.x;
  if (gid < total) dst[gid] = f2bf(src[gid]);
}

// ---------------- input projection (K = Fe+1 = 9, mask channel == 1) ------------------
__global__ __launch_bounds__(256) void k_proj(const float* __restrict__ X,
    const float* __restrict__ Wp, const float* __restrict__ bp,
    unsigned short* __restrict__ dst){
  long gid = (long)blockIdx.x*256 + threadIdx.x;
  long row = gid >> 6; int h = (int)(gid & 63);
  float s = bp[h] + Wp[FE*HH + h];
  const float* xr = X + row*FE;
#pragma unroll
  for (int f=0; f<FE; ++f) s += xr[f]*Wp[f*HH + h];
  dst[gid] = f2bf(s);
}

// ---------------- gated causal temporal conv via 5 shifted WMMA GEMMs -----------------
__global__ __launch_bounds__(128) void k_gated_conv(const unsigned short* __restrict__ src,
    const unsigned short* __restrict__ wct, const unsigned short* __restrict__ wgt,
    const float* __restrict__ bc, const float* __restrict__ bg,
    float* __restrict__ o0, float* __restrict__ o1, unsigned short* __restrict__ o2,
    int mode){
  const int lane = threadIdx.x & 31;
  const int wv   = threadIdx.x >> 5;
  const int g    = lane >> 4;
  const int lr   = lane & 15;
  int rowBase = blockIdx.x*64 + wv*16;
  int b  = rowBase / (TT*NN);
  int tn = rowBase - b*(TT*NN);
  int t  = tn / NN;
  int nb = tn - t*NN;
  FragC ca[4], ga[4];
#pragma unroll
  for (int i=0;i<4;++i)
#pragma unroll
    for (int e=0;e<8;++e){ ca[i].f[e]=0.f; ga[i].f[e]=0.f; }

  for (int k=0;k<KS;++k){
    int ts = t + k - (KS-1);
    if (ts < 0) continue;
    const unsigned short* abase = src + (((long)(b*TT + ts)*NN + nb)*HH);
#pragma unroll
    for (int ch=0; ch<HH; ch+=32){
      FragAB a; load_frag(a, abase, lr, HH, ch, g);
#pragma unroll
      for (int tl=0; tl<4; ++tl){
        FragAB wc; load_frag(wc, wct + k*HH*HH, tl*16+lr, HH, ch, g);
        ca[tl].v = wmma_bf16(a, wc, ca[tl].v);
        FragAB wg; load_frag(wg, wgt + k*HH*HH, tl*16+lr, HH, ch, g);
        ga[tl].v = wmma_bf16(a, wg, ga[tl].v);
      }
    }
  }
#pragma unroll
  for (int tl=0; tl<4; ++tl){
    int ho = tl*16 + lr;
    float bcv = bc[ho], bgv = bg[ho];
#pragma unroll
    for (int e=0;e<8;++e){
      int n = nb + 8*g + e;
      float cv = ca[tl].f[e] + bcv;
      float gv = ga[tl].f[e] + bgv;
      float y  = tanhf(cv) * (1.f/(1.f + __expf(-gv)));
      if (mode == 0){
        long idx = ((long)(b*TT + t)*NN + n)*HH + ho;
        o0[idx] = y;
        o1[idx] = y;
        o2[((long)(b*TT + t)*HH + ho)*NN + n] = f2bf(y);
      } else if (mode == 1){
        o0[(long)(b*NN + n)*(2*THd) + t*HH + ho] = y;
        o1[(long)(b*NN + n)*THd + t*HH + ho]     = y;
      } else {
        o0[(long)(b*NN + n)*(2*THd) + THd + t*HH + ho] = y;
      }
    }
  }
}

// ---------------- Q/K projections: Xt[b,n,1536] @ Wq/Wk (from transposed x1) ----------
__global__ __launch_bounds__(128) void k_qk(const unsigned short* __restrict__ srcT,
    const unsigned short* __restrict__ wqt, const unsigned short* __restrict__ wkt,
    const float* __restrict__ bq, const float* __restrict__ bk,
    unsigned short* __restrict__ Qb, unsigned short* __restrict__ Kb){
  const int lane = threadIdx.x & 31, wv = threadIdx.x >> 5;
  const int g = lane >> 4, lr = lane & 15;
  int b  = blockIdx.x / (NN/64);
  int nb = (blockIdx.x % (NN/64))*64 + wv*16;
  int n  = nb + lr;
  const unsigned short* xb = srcT + (long)b*TT*HH*NN;
  FragC aq[4], ak[4];
#pragma unroll
  for (int i=0;i<4;++i)
#pragma unroll
    for (int e=0;e<8;++e){ aq[i].f[e]=0.f; ak[i].f[e]=0.f; }

  for (int ch=0; ch<THd; ch+=32){
    FragAB a;
#pragma unroll
    for (int p=0;p<8;++p){
      int k0 = ch + ((p<4) ? 2*p : 16 + 2*(p-4)) + 8*g;
      int t0 = k0 >> 6, h0 = k0 & 63;
      const unsigned short* qp = xb + ((long)t0*HH + h0)*NN + n;
      a.u[2*p]   = qp[0];
      a.u[2*p+1] = qp[NN];
    }
#pragma unroll
    for (int tl=0; tl<4; ++tl){
      FragAB w1; load_frag(w1, wqt, tl*16+lr, THd, ch, g);
      aq[tl].v = wmma_bf16(a, w1, aq[tl].v);
      FragAB w2; load_frag(w2, wkt, tl*16+lr, THd, ch, g);
      ak[tl].v = wmma_bf16(a, w2, ak[tl].v);
    }
  }
#pragma unroll
  for (int tl=0; tl<4; ++tl){
    int col = tl*16 + lr;
    float bqv = bq[col], bkv = bk[col];
#pragma unroll
    for (int e=0;e<8;++e){
      int nn2 = nb + 8*g + e;
      Qb[((long)b*NN + nn2)*HH + col] = f2bf(aq[tl].f[e] + bqv);
      Kb[((long)b*NN + nn2)*HH + col] = f2bf(ak[tl].f[e] + bkv);
    }
  }
}

// ---------------- scores = Q @ K^T / 8, adjacency mask fused --------------------------
__global__ __launch_bounds__(128) void k_scores(const unsigned short* __restrict__ Qb,
    const unsigned short* __restrict__ Kb, const int* __restrict__ adj,
    unsigned short* __restrict__ Pb){
  const int lane = threadIdx.x & 31, wv = threadIdx.x >> 5;
  const int g = lane >> 4, lr = lane & 15;
  int b  = blockIdx.x >> 8;
  int rc = blockIdx.x & 255;
  int nb = (rc >> 4)*64 + wv*16;
  int mb = (rc & 15)*64;
  const unsigned short* qa = Qb + ((long)b*NN + nb)*HH;
  const unsigned short* ka = Kb + (long)b*NN*HH;
  FragC acc[4];
#pragma unroll
  for (int i=0;i<4;++i)
#pragma unroll
    for (int e=0;e<8;++e) acc[i].f[e]=0.f;
#pragma unroll
  for (int ch=0; ch<HH; ch+=32){
    FragAB a; load_frag(a, qa, lr, HH, ch, g);
#pragma unroll
    for (int tl=0; tl<4; ++tl){
      FragAB kf; load_frag(kf, ka, mb + tl*16 + lr, HH, ch, g);
      acc[tl].v = wmma_bf16(a, kf, acc[tl].v);
    }
  }
#pragma unroll
  for (int tl=0; tl<4; ++tl){
    int m = mb + tl*16 + lr;
#pragma unroll
    for (int e=0;e<8;++e){
      int n = nb + 8*g + e;
      float raw = acc[tl].f[e]*0.125f;
      bool conn = (n==m) || (adj[(long)n*NN + m] != 0);
      Pb[((long)b*NN + n)*NN + m] = conn ? f2bf(raw) : (unsigned short)0xFF80;
    }
  }
}

// ---------------- row softmax in place on bf16 scores ---------------------------------
__global__ __launch_bounds__(256) void k_softmax(unsigned short* __restrict__ Pb){
  __shared__ float red[256];
  long row = blockIdx.x;
  unsigned short* p = Pb + row*NN;
  int tid = threadIdx.x;
  float v[4];
#pragma unroll
  for (int j=0;j<4;++j) v[j] = bf2f(p[tid*4 + j]);
  float mx = fmaxf(fmaxf(v[0],v[1]), fmaxf(v[2],v[3]));
  red[tid] = mx; __syncthreads();
  for (int s=128; s>0; s>>=1){ if (tid<s) red[tid] = fmaxf(red[tid], red[tid+s]); __syncthreads(); }
  mx = red[0]; __syncthreads();
  float sm = 0.f;
#pragma unroll
  for (int j=0;j<4;++j){ v[j] = __expf(v[j]-mx); sm += v[j]; }
  red[tid] = sm; __syncthreads();
  for (int s=128; s>0; s>>=1){ if (tid<s) red[tid] += red[tid+s]; __syncthreads(); }
  float inv = 1.f/red[0];
#pragma unroll
  for (int j=0;j<4;++j) p[tid*4 + j] = f2bf(v[j]*inv);
}

// ---------------- diffusion hop: v' = P @ v, TDM-staged B operand in LDS --------------
// B tile [64 h][1024 m] is shared by all 4 waves: the TDM DMAs K-chunks of 128 into a
// double-buffered, TDM-padded LDS tile (row stride 272B) while WMMAs consume the
// previous chunk. A (P rows) stays in global/L2 (reused 24x across t).
__global__ __launch_bounds__(128) void k_hop(const unsigned short* __restrict__ Pb,
    const unsigned short* __restrict__ vin, unsigned short* __restrict__ vout,
    float* __restrict__ fsum, float* __restrict__ bsum,
    const float* __restrict__ aF, const float* __restrict__ aB, int hop){
  __shared__ unsigned short shb[2][HH*LDP];
  const int lane = threadIdx.x & 31, wv = threadIdx.x >> 5;
  const int g = lane >> 4, lr = lane & 15;
  int blk = blockIdx.x;
  int b   = blk / (TT*16);
  int rem = blk - b*(TT*16);
  int t   = rem >> 4;
  int nb  = (rem & 15)*64 + wv*16;
  const unsigned short* pa    = Pb  + ((long)b*NN + nb)*NN;
  const unsigned short* vbase = vin + (long)(b*TT + t)*HH*NN;
  unsigned lds0 = (unsigned)(unsigned long long)(&shb[0][0]);
  unsigned lds1 = (unsigned)(unsigned long long)(&shb[1][0]);

  FragC acc[4];
#pragma unroll
  for (int i=0;i<4;++i)
#pragma unroll
    for (int e=0;e<8;++e) acc[i].f[e]=0.f;

  if (threadIdx.x < 32){
    tdm_load_tile(lds0, vbase);                    // chunk 0
    __builtin_amdgcn_s_wait_tensorcnt(0);
  }
  __syncthreads();

  for (int c=0; c<NN/KC; ++c){
    if (c+1 < NN/KC && threadIdx.x < 32)
      tdm_load_tile((c&1) ? lds0 : lds1, vbase + (c+1)*KC);   // async next chunk

    const unsigned short* bs = shb[c&1];
    // hoist the 4 A fragments of this chunk so global loads overlap LDS reads
    FragAB afr[4];
#pragma unroll
    for (int s=0;s<4;++s) load_frag(afr[s], pa, lr, NN, c*KC + s*32, g);
    if (c+1 < NN/KC) __builtin_prefetch(pa + (long)lr*NN + (c+1)*KC, 0, 0);
#pragma unroll
    for (int s=0;s<4;++s){
#pragma unroll
      for (int tl=0; tl<4; ++tl){
        FragAB bf; load_frag(bf, bs, tl*16+lr, LDP, s*32, g);
        acc[tl].v = wmma_bf16(afr[s], bf, acc[tl].v);
      }
    }
    __syncthreads();                               // all waves done reading shb[c&1]
    if (c+1 < NN/KC){
      if (threadIdx.x < 32) __builtin_amdgcn_s_wait_tensorcnt(0);
      __syncthreads();                             // next chunk visible to all waves
    }
  }

  float af = aF[hop], ab = aB[hop];
  long base = (long)(b*TT + t);
#pragma unroll
  for (int tl=0; tl<4; ++tl){
    int h = tl*16 + lr;
    union { unsigned short s[8]; uint4 q; } pk;
#pragma unroll
    for (int e=0;e<8;++e){
      int n = nb + 8*g + e;
      float val = acc[tl].f[e];
      long idx = (base*NN + n)*HH + h;
      fsum[idx] += af*val;
      bsum[idx] += ab*val;
      pk.s[e] = f2bf(val);
    }
    *(uint4*)(vout + (base*HH + h)*NN + nb + 8*g) = pk.q;
  }
}

// ---------------- tanh(A[M,64] @ W[64,64] + b) -> bf16 --------------------------------
__global__ __launch_bounds__(128) void k_gemm64(const unsigned short* __restrict__ src,
    const unsigned short* __restrict__ wt, const float* __restrict__ bias,
    unsigned short* __restrict__ dst){
  const int lane = threadIdx.x & 31, wv = threadIdx.x >> 5;
  const int g = lane >> 4, lr = lane & 15;
  long rowBase = (long)blockIdx.x*64 + wv*16;
  const unsigned short* ab = src + rowBase*HH;
  FragC acc[4];
#pragma unroll
  for (int i=0;i<4;++i)
#pragma unroll
    for (int e=0;e<8;++e) acc[i].f[e]=0.f;
#pragma unroll
  for (int ch=0; ch<HH; ch+=32){
    FragAB a; load_frag(a, ab, lr, HH, ch, g);
#pragma unroll
    for (int tl=0; tl<4; ++tl){
      FragAB w; load_frag(w, wt, tl*16+lr, HH, ch, g);
      acc[tl].v = wmma_bf16(a, w, acc[tl].v);
    }
  }
#pragma unroll
  for (int tl=0; tl<4; ++tl){
    int col = tl*16 + lr; float bv = bias[col];
#pragma unroll
    for (int e=0;e<8;++e){
      long r = rowBase + 8*g + e;
      dst[r*HH + col] = f2bf(tanhf(acc[tl].f[e] + bv));
    }
  }
}

// =======================================================================================
extern "C" void kernel_launch(void* const* d_in, const int* in_sizes, int n_in,
                              void* d_out, int out_size, void* d_ws, size_t ws_size,
                              hipStream_t stream) {
  const float* X   = (const float*)d_in[0];
  const int*   adj = (const int*)  d_in[1];
  const float* Wp  = (const float*)d_in[3];  const float* bp  = (const float*)d_in[4];
  const float* t1wc= (const float*)d_in[5];  const float* t1bc= (const float*)d_in[6];
  const float* t1wg= (const float*)d_in[7];  const float* t1bg= (const float*)d_in[8];
  const float* Wq  = (const float*)d_in[9];  const float* bq  = (const float*)d_in[10];
  const float* Wk  = (const float*)d_in[11]; const float* bk  = (const float*)d_in[12];
  const float* aF  = (const float*)d_in[13]; const float* aB  = (const float*)d_in[14];
  const float* Wpf = (const float*)d_in[15]; const float* bpf = (const float*)d_in[16];
  const float* Wpb = (const float*)d_in[17]; const float* bpb = (const float*)d_in[18];
  const float* tfwc= (const float*)d_in[19]; const float* tfbc= (const float*)d_in[20];
  const float* tfwg= (const float*)d_in[21]; const float* tfbg= (const float*)d_in[22];
  const float* tbwc= (const float*)d_in[23]; const float* tbbc= (const float*)d_in[24];
  const float* tbwg= (const float*)d_in[25]; const float* tbbg= (const float*)d_in[26];

  char* ws = (char*)d_ws;
  unsigned short* buf1 = (unsigned short*)(ws);                 // x0b -> fsb -> Bwdb
  unsigned short* buf2 = (unsigned short*)(ws +  25165824L);    // x1bT -> Fwdb
  unsigned short* buf3 = (unsigned short*)(ws +  50331648L);    // v ping -> bsb
  float*          fsum = (float*)         (ws +  75497472L);
  float*          bsum = (float*)         (ws + 125829120L);
  unsigned short* Qb   = (unsigned short*)(ws + 176160768L);
  unsigned short* Kb   = (unsigned short*)(ws + 177209344L);
  unsigned short* Pb   = (unsigned short*)(ws + 178257920L);
  unsigned short* wts  = (unsigned short*)(ws + 195035136L);

  unsigned short* w_t1c = wts;
  unsigned short* w_t1g = wts +  20480;
  unsigned short* w_q   = wts +  40960;
  unsigned short* w_k   = wts + 139264;
  unsigned short* w_pf  = wts + 237568;
  unsigned short* w_pb  = wts + 241664;
  unsigned short* w_tfc = wts + 245760;
  unsigned short* w_tfg = wts + 266240;
  unsigned short* w_tbc = wts + 286720;
  unsigned short* w_tbg = wts + 307200;

  k_convert_tr<<<80, 256, 0, stream>>>(t1wc, w_t1c,   64, 20480);
  k_convert_tr<<<80, 256, 0, stream>>>(t1wg, w_t1g,   64, 20480);
  k_convert_tr<<<384,256, 0, stream>>>(Wq,   w_q,   1536, 98304);
  k_convert_tr<<<384,256, 0, stream>>>(Wk,   w_k,   1536, 98304);
  k_convert_tr<<<16, 256, 0, stream>>>(Wpf,  w_pf,    64,  4096);
  k_convert_tr<<<16, 256, 0, stream>>>(Wpb,  w_pb,    64,  4096);
  k_convert_tr<<<80, 256, 0, stream>>>(tfwc, w_tfc,   64, 20480);
  k_convert_tr<<<80, 256, 0, stream>>>(tfwg, w_tfg,   64, 20480);
  k_convert_tr<<<80, 256, 0, stream>>>(tbwc, w_tbc,   64, 20480);
  k_convert_tr<<<80, 256, 0, stream>>>(tbwg, w_tbg,   64, 20480);

  k_proj<<<49152, 256, 0, stream>>>(X, Wp, bp, buf1);

  k_gated_conv<<<3072, 128, 0, stream>>>(buf1, w_t1c, w_t1g, t1bc, t1bg,
                                         fsum, bsum, buf2, 0);

  k_qk<<<128, 128, 0, stream>>>(buf2, w_q, w_k, bq, bk, Qb, Kb);
  k_scores<<<2048, 128, 0, stream>>>(Qb, Kb, adj, Pb);
  k_softmax<<<8192, 256, 0, stream>>>(Pb);

  k_hop<<<3072, 128, 0, stream>>>(Pb, buf2, buf3, fsum, bsum, aF, aB, 0);
  k_hop<<<3072, 128, 0, stream>>>(Pb, buf3, buf2, fsum, bsum, aF, aB, 1);
  k_hop<<<3072, 128, 0, stream>>>(Pb, buf2, buf3, fsum, bsum, aF, aB, 2);

  k_convert<<<49152, 256, 0, stream>>>(fsum, buf1, SZ);
  k_convert<<<49152, 256, 0, stream>>>(bsum, buf3, SZ);

  k_gemm64<<<3072, 128, 0, stream>>>(buf1, w_pf, bpf, buf2);   // Fwdb -> buf2
  k_gemm64<<<3072, 128, 0, stream>>>(buf3, w_pb, bpb, buf1);   // Bwdb -> buf1

  float* out  = (float*)d_out;
  float* out2 = out + (long)BB*NN*2*THd;
  k_gated_conv<<<3072, 128, 0, stream>>>(buf2, w_tfc, w_tfg, tfbc, tfbg,
                                         out, out2, nullptr, 1);
  k_gated_conv<<<3072, 128, 0, stream>>>(buf1, w_tbc, w_tbg, tbbc, tbbg,
                                         out, nullptr, nullptr, 2);
}
